// GCK3x3Layer_10222022165341
// MI455X (gfx1250) — matrix operand order repaired and checked by reading, not compiled, not run on hardware
//
#include <hip/hip_runtime.h>

// ---------------- types ----------------
typedef __bf16 bf16_t;
typedef __attribute__((ext_vector_type(16))) __bf16 v16bf;
typedef __attribute__((ext_vector_type(8)))  float  v8f;
typedef __attribute__((ext_vector_type(4)))  int    v4i;

#define IN_C   256
#define OUT_C  256
#define H      258
#define W      258
#define HO     256
#define WO     256
#define INHW   (H * W)                 // 66564
#define OUTHW  (HO * WO)               // 65536
#define RCNT   9                       // 3x3 taps
#define NIT    (RCNT * (IN_C / 32))    // 72 k-steps of 32
#define BN     64                      // workgroup spatial tile (n)
#define LDSS   40                      // padded LDS row stride (elements)

union FragBF { v16bf v; v4i q[2]; };
union Pack8  { bf16_t e[8]; v4i q; };

// -------- prep: split fp32 weights into bf16 hi/lo planes [r][oc][ic] --------
__global__ __launch_bounds__(256)
void prep_weights(const float* __restrict__ w,
                  unsigned short* __restrict__ whi,
                  unsigned short* __restrict__ wlo) {
    int idx = blockIdx.x * 256 + threadIdx.x;      // 256*2304 total, exact grid
    if (idx >= OUT_C * IN_C * RCNT) return;
    int oc = idx / (IN_C * RCNT);
    int k  = idx % (IN_C * RCNT);
    int ic = k / RCNT;
    int r  = k % RCNT;
    float  x  = w[idx];
    bf16_t h  = (bf16_t)x;
    float  rl = x - (float)h;
    bf16_t l  = (bf16_t)rl;
    int o = (r * OUT_C + oc) * IN_C + ic;
    whi[o] = __builtin_bit_cast(unsigned short, h);
    wlo[o] = __builtin_bit_cast(unsigned short, l);
}

// -------- main: 9 accumulated GEMMs, bf16x3 split precision WMMA --------
__global__ __launch_bounds__(256)
void conv3x3_wmma(const float* __restrict__ in,
                  const unsigned short* __restrict__ whi_u,
                  const unsigned short* __restrict__ wlo_u,
                  float* __restrict__ out) {
    const bf16_t* whi = (const bf16_t*)whi_u;
    const bf16_t* wlo = (const bf16_t*)wlo_u;

    __shared__ __attribute__((aligned(16))) bf16_t Bhi[2][BN * LDSS];
    __shared__ __attribute__((aligned(16))) bf16_t Blo[2][BN * LDSS];

    const int tid  = threadIdx.x;
    const int lane = tid & 31;
    const int wav  = tid >> 5;
    const int lm   = lane & 15;      // column / row index within fragment
    const int lh   = lane >> 4;      // half-wave select

    const int oh  = blockIdx.y;
    const int ow0 = blockIdx.x * BN;
    const int m0  = blockIdx.z * 128 + (wav >> 1) * 32;  // wave's oc base
    const int n0  = (wav & 1) * 32;                      // wave's local n base

    // B-tile loader mapping: 256 threads cover 64 n x 32 k (8 k each)
    const int ln  = tid & 63;        // n within tile
    const int lkq = tid >> 6;        // ic sub-block of 8 (0..3)

    v8f acc[2][2] = {};

    // per-lane A row offsets inside a 256x256 plane (element units)
    // A k order used here (contiguous, matches B): lane<16 -> K0..15; lane>=16 -> K16..31
    int rowA[2];
    rowA[0] = (m0 +  0 + lm) * IN_C + lh * 16;
    rowA[1] = (m0 + 16 + lm) * IN_C + lh * 16;

    // ---- stage tile 0 (r=0 -> kh=0,kw=0; ic0=0) ----
    {
        const float* src = in + (size_t)(lkq * 8) * INHW + (size_t)oh * W + (ow0 + ln);
        Pack8 ph, pl;
        #pragma unroll
        for (int j = 0; j < 8; ++j) {
            float  x  = src[(size_t)j * INHW];
            bf16_t h  = (bf16_t)x;
            float  rl = x - (float)h;
            ph.e[j] = h;
            pl.e[j] = (bf16_t)rl;
        }
        *(v4i*)&Bhi[0][ln * LDSS + lkq * 8] = ph.q;
        *(v4i*)&Blo[0][ln * LDSS + lkq * 8] = pl.q;
    }
    __syncthreads();

    #pragma unroll 1
    for (int it = 0; it < NIT; ++it) {
        const int cur = it & 1;

        // ---- prefetch next B tile into registers (overlaps with WMMA) ----
        float vals[8];
        if (it + 1 < NIT) {
            const int itn = it + 1;
            const int r   = itn >> 3;
            const int ic0 = (itn & 7) * 32;
            const int kh  = r / 3, kw = r % 3;
            const float* src = in + (size_t)(ic0 + lkq * 8) * INHW
                                  + (size_t)(oh + kh) * W + (ow0 + kw + ln);
            #pragma unroll
            for (int j = 0; j < 8; ++j) vals[j] = src[(size_t)j * INHW];
        }

        // ---- compute current k-step ----
        {
            const int r   = it >> 3;
            const int ic0 = (it & 7) * 32;
            const size_t planeoff = (size_t)r * (OUT_C * IN_C) + ic0;

            FragBF ahi[2], alo[2], bhi[2], blo[2];
            #pragma unroll
            for (int mi = 0; mi < 2; ++mi) {
                const bf16_t* ph = whi + planeoff + rowA[mi];
                const bf16_t* pl = wlo + planeoff + rowA[mi];
                ahi[mi].q[0] = *(const v4i*)(ph);
                ahi[mi].q[1] = *(const v4i*)(ph + 8);   // K+8 group
                alo[mi].q[0] = *(const v4i*)(pl);
                alo[mi].q[1] = *(const v4i*)(pl + 8);
            }
            // B layout: lane<16 -> K0..15 of col n; lane>=16 -> K16..31
            #pragma unroll
            for (int ni = 0; ni < 2; ++ni) {
                const int nn = n0 + ni * 16 + lm;
                const int bo = nn * LDSS + lh * 16;
                bhi[ni].q[0] = *(const v4i*)&Bhi[cur][bo];
                bhi[ni].q[1] = *(const v4i*)&Bhi[cur][bo + 8];
                blo[ni].q[0] = *(const v4i*)&Blo[cur][bo];
                blo[ni].q[1] = *(const v4i*)&Blo[cur][bo + 8];
            }
            #pragma unroll
            for (int mi = 0; mi < 2; ++mi) {
                #pragma unroll
                for (int ni = 0; ni < 2; ++ni) {
                    acc[mi][ni] = __builtin_amdgcn_wmma_f32_16x16x32_bf16(
                        false, ahi[mi].v, false, bhi[ni].v, (short)0, acc[mi][ni], false, false);
                    acc[mi][ni] = __builtin_amdgcn_wmma_f32_16x16x32_bf16(
                        false, ahi[mi].v, false, blo[ni].v, (short)0, acc[mi][ni], false, false);
                    acc[mi][ni] = __builtin_amdgcn_wmma_f32_16x16x32_bf16(
                        false, alo[mi].v, false, bhi[ni].v, (short)0, acc[mi][ni], false, false);
                }
            }
        }

        // ---- split + store next tile into the other LDS buffer ----
        if (it + 1 < NIT) {
            const int nb = (it + 1) & 1;
            Pack8 ph, pl;
            #pragma unroll
            for (int j = 0; j < 8; ++j) {
                float  x  = vals[j];
                bf16_t h  = (bf16_t)x;
                float  rl = x - (float)h;
                ph.e[j] = h;
                pl.e[j] = (bf16_t)rl;
            }
            *(v4i*)&Bhi[nb][ln * LDSS + lkq * 8] = ph.q;
            *(v4i*)&Blo[nb][ln * LDSS + lkq * 8] = pl.q;
        }
        __syncthreads();
    }

    // ---- epilogue: D layout -> VGPR v holds M = v + lh*8, N = lm ----
    #pragma unroll
    for (int mi = 0; mi < 2; ++mi) {
        #pragma unroll
        for (int ni = 0; ni < 2; ++ni) {
            const int oc = m0 + mi * 16 + lh * 8;
            const int ow = ow0 + n0 + ni * 16 + lm;
            float* p = out + (size_t)oc * OUTHW + (size_t)oh * WO + ow;
            #pragma unroll
            for (int v = 0; v < 8; ++v)
                p[(size_t)v * OUTHW] = acc[mi][ni][v];
        }
    }
}

extern "C" void kernel_launch(void* const* d_in, const int* in_sizes, int n_in,
                              void* d_out, int out_size, void* d_ws, size_t ws_size,
                              hipStream_t stream) {
    const float* inp = (const float*)d_in[0];   // (1,256,258,258) fp32
    const float* wts = (const float*)d_in[1];   // (256, 2304) fp32
    float* out = (float*)d_out;                 // (1,256,256,256) fp32

    // workspace: hi plane then lo plane, each 9*256*256 bf16 (~1.18 MB)
    unsigned short* whi = (unsigned short*)d_ws;
    unsigned short* wlo = whi + (size_t)RCNT * OUT_C * IN_C;

    prep_weights<<<(OUT_C * IN_C * RCNT) / 256, 256, 0, stream>>>(wts, whi, wlo);

    dim3 grid(WO / BN, HO, OUT_C / 128);        // (4, 256, 2)
    conv3x3_wmma<<<grid, 256, 0, stream>>>(inp, whi, wlo, out);
}
